// TemporalVisualCoherenceModule_85272280695284
// MI455X (gfx1250) — matrix operand order repaired and checked by reading, not compiled, not run on hardware
//
#include <hip/hip_runtime.h>
#include <stdint.h>

#define D     1024
#define TT    256
#define BB    8
#define NN    2048            // B*T
#define HM    16
#define DH    64
#define HG    8
#define DG    128
#define KCONV 7

typedef __bf16 v16bf __attribute__((ext_vector_type(16)));
typedef float  v8f   __attribute__((ext_vector_type(8)));

// ---------------------------------------------------------------- bf16 utils
__device__ __forceinline__ unsigned short f2bf(float f) {
  union { float f; uint32_t u; } c; c.f = f;
  uint32_t u = c.u;
  u += 0x7FFFu + ((u >> 16) & 1u);         // round-to-nearest-even
  return (unsigned short)(u >> 16);
}
__device__ __forceinline__ float bf2f(unsigned short h) {
  union { float f; uint32_t u; } c; c.u = ((uint32_t)h) << 16; return c.f;
}
__device__ __forceinline__ uint32_t pack2bf(float a, float b) {
  return (uint32_t)f2bf(a) | ((uint32_t)f2bf(b) << 16);
}
__device__ __forceinline__ float lrelu(float x) { return x > 0.f ? x : 0.2f * x; }

// ---------------------------------------------------------------- WMMA core
__device__ __forceinline__ v8f wmma_bf16(v16bf a, v16bf b, v8f c) {
  // D = A(16x32 bf16) * B(32x16 bf16) + C(16x16 f32)
  return __builtin_amdgcn_wmma_f32_16x16x32_bf16(false, a, false, b, (short)0, c,
                                                 false, false);
}

// --------------------------------------------- Tensor Data Mover (optional)
// 2D tile DMA global->LDS per CDNA5 ISA ch.7/8 (VIMAGE TENSOR_LOAD_TO_LDS).
// Probe-verified builtin: ROCm 7.2 (clang-22) = 5 args, amdgpu-toolchain
// (clang-23) = 6 args. Guarded; falls back to direct global loads if absent.
#if defined(__has_builtin)
#if __has_builtin(__builtin_amdgcn_tensor_load_to_lds) && \
    __has_builtin(__builtin_amdgcn_s_wait_tensorcnt)
#define USE_TDM 1
#endif
#endif

typedef uint32_t u32x4 __attribute__((ext_vector_type(4)));
typedef int      i32x4 __attribute__((ext_vector_type(4)));
typedef int      i32x8 __attribute__((ext_vector_type(8)));

#ifdef USE_TDM
// Build D# for a dense 2D tile: tile_w (bf16 elems) x tile_h rows, row stride
// = lda elems, destination LDS byte offset lds_off.
__device__ __forceinline__ void tdm_load_2d(uint32_t lds_off, const void* gsrc,
                                            uint32_t lda_elems, uint32_t tile_w,
                                            uint32_t tile_h) {
  uint64_t ga = (uint64_t)(uintptr_t)gsrc;
  u32x4 g0;
  g0.x = 1u;                                   // count=1 (valid), user mode
  g0.y = lds_off;                              // D#.lds_addr (bytes)
  g0.z = (uint32_t)ga;                         // global_addr[31:0]
  g0.w = (uint32_t)(ga >> 32) | 0x80000000u;   // global_addr[56:32] | type=2
  i32x8 g1;
  g1[0] = 0x10000;                             // data_size=1 (2B), wg_mask=0
  g1[1] = (int)((lda_elems & 0xFFFFu) << 16);  // tensor_dim0[15:0]
  g1[2] = (int)((lda_elems >> 16) | 0xFFFF0000u); // dim0[31:16] | dim1[15:0]=big
  g1[3] = (int)(tile_w << 16);                 // tensor_dim1 hi=0 | tile_dim0
  g1[4] = (int)tile_h;                         // tile_dim1 | tile_dim2=0
  g1[5] = (int)lda_elems;                      // tensor_dim0_stride[31:0]
  g1[6] = 0;                                   // stride hi | dim1_stride lo
  g1[7] = 0;
  i32x4 gz = {0, 0, 0, 0};
#if defined(__clang_major__) && __clang_major__ >= 23
  i32x8 gz8 = {0, 0, 0, 0, 0, 0, 0, 0};
  __builtin_amdgcn_tensor_load_to_lds(g0, g1, gz, gz, gz8, 0);
#else
  __builtin_amdgcn_tensor_load_to_lds(g0, g1, gz, gz, 0);
#endif
}
#endif

// A fragment: 16x32 bf16, row-major source. lane<16: M=lane, K groups {0..7,16..23};
// lane>=16: K groups {8..15,24..31}.  => two 16-byte contiguous loads per lane.
__device__ __forceinline__ v16bf load_a_frag(const unsigned short* __restrict__ A,
                                             int lda, int m0, int k0) {
  int lane = threadIdx.x & 31;
  int m = lane & 15, g = lane >> 4;
  const unsigned short* p = A + (size_t)(m0 + m) * lda + k0 + 8 * g;
  union { v16bf v; uint4 u[2]; } r;
  r.u[0] = *reinterpret_cast<const uint4*>(p);
  r.u[1] = *reinterpret_cast<const uint4*>(p + 16);
  return r.v;
}

// A fragment with temporal shift + zero predication (dilated conv taps).
__device__ __forceinline__ v16bf load_a_frag_shift(const unsigned short* __restrict__ X,
                                                   int m0, int k0, int shift) {
  int lane = threadIdx.x & 31;
  int m = lane & 15, g = lane >> 4;
  int row = m0 + m;
  int t = (row & (TT - 1)) + shift;
  union { v16bf v; uint4 u[2]; } r;
  if (t >= 0 && t < TT) {
    const unsigned short* p = X + (size_t)(row + shift) * D + k0 + 8 * g;
    r.u[0] = *reinterpret_cast<const uint4*>(p);
    r.u[1] = *reinterpret_cast<const uint4*>(p + 16);
  } else {
    r.u[0] = make_uint4(0u, 0u, 0u, 0u);
    r.u[1] = make_uint4(0u, 0u, 0u, 0u);
  }
  return r.v;
}

// B fragment, BT layout: B = W^T with W row-major (out x K). lane: n=lane&15,
// h=lane>>4 -> 8 bf16 at W[n0+n][k0+8h] and 8 at W[n0+n][k0+16+8h].
__device__ __forceinline__ v16bf load_bt_frag(const unsigned short* __restrict__ W,
                                              int ldb, int n0, int k0) {
  int lane = threadIdx.x & 31;
  int n = lane & 15, h = lane >> 4;
  const unsigned short* p = W + (size_t)(n0 + n) * ldb + k0 + 8 * h;
  union { v16bf v; uint4 u[2]; } r;
  r.u[0] = *reinterpret_cast<const uint4*>(p);
  r.u[1] = *reinterpret_cast<const uint4*>(p + 16);
  return r.v;
}

// B fragment, BN layout: B row-major (K x N); strided 16-bit gathers.
__device__ __forceinline__ v16bf load_bn_frag(const unsigned short* __restrict__ Bm,
                                              int ldb, int n0, int k0) {
  int lane = threadIdx.x & 31;
  int n = lane & 15, h = lane >> 4;
  const unsigned short* p = Bm + (size_t)(k0 + 8 * h) * ldb + n0 + n;
  union { v16bf v; unsigned short s[16]; } r;
#pragma unroll
  for (int i = 0; i < 8; ++i) {
    r.s[2 * i]     = p[(size_t)(2 * i + (i >= 4 ? 8 : 0)) * ldb];
    r.s[2 * i + 1] = p[(size_t)(2 * i + 1 + (i >= 4 ? 8 : 0)) * ldb];
  }
  return r.v;
}

// C/D 16x16 f32 store: lane n=lane&15, rows m = r + 8*(lane>>4).
__device__ __forceinline__ void store_tile(v8f acc, float* Cf, unsigned short* Cb,
                                           int ldc, int m0, int n0,
                                           const float* bias, float alpha) {
  int lane = threadIdx.x & 31;
  int n = lane & 15, half = lane >> 4;
  float b = bias ? bias[n0 + n] : 0.f;
#pragma unroll
  for (int r = 0; r < 8; ++r) {
    int m = m0 + r + 8 * half;
    float v = acc[r] * alpha + b;
    if (Cf) Cf[(size_t)m * ldc + n0 + n] = v;
    if (Cb) Cb[(size_t)m * ldc + n0 + n] = f2bf(v);
  }
}

// ------------------------------------------------- generic batched WMMA GEMM
// C[z] = A[z] (rows x K, bf16) * op(B[z]) * alpha + bias
// BT=true:  B = W^T, W row-major (Ncols x K).  BT=false: B row-major (K x Ncols).
// Each wave owns a 16 x (16*NACC) strip; all waves in a block share the same
// 16 output rows, so the A tile is staged once per K-step into LDS by the
// Tensor Data Mover (double-buffered, s_wait_tensorcnt) and read back as
// fragments via ds_load. Falls back to direct global A loads without TDM.
template <bool BT, int NACC>
__global__ void gemm_wmma(const unsigned short* __restrict__ A, int lda,
                          const unsigned short* __restrict__ Bm, int ldb,
                          float* Cf, unsigned short* Cb, int ldc,
                          const float* __restrict__ bias, float alpha,
                          int Ksize, int Ncols, int zdiv,
                          long sA1, long sA2, long sB1, long sB2, long sC1, long sC2) {
  int z = blockIdx.z;
  int z1 = z / zdiv, z2 = z % zdiv;
  A += z1 * sA1 + z2 * sA2;
  Bm += z1 * sB1 + z2 * sB2;
  long coff = z1 * sC1 + z2 * sC2;
  if (Cf) Cf += coff;
  if (Cb) Cb += coff;

  int wave = threadIdx.x >> 5;
  int m0 = blockIdx.y * 16;
  int n0 = (blockIdx.x * 4 + wave) * (16 * NACC);
  bool active = (n0 < Ncols);        // keep inactive waves in barrier schedule

  v8f acc[NACC];
#pragma unroll
  for (int c = 0; c < NACC; ++c) acc[c] = v8f{};

#ifdef USE_TDM
  __shared__ __attribute__((aligned(16))) unsigned short atile[2][16 * 32];
  const int nsteps = Ksize / 32;
  const bool issuer = (threadIdx.x >> 5) == 0;   // wave 0 drives the TDM
  if (issuer)
    tdm_load_2d((uint32_t)(uintptr_t)&atile[0][0], A + (size_t)m0 * lda,
                (uint32_t)lda, 32u, 16u);
  for (int s = 0; s < nsteps; ++s) {
    int k0 = s * 32;
    int p = s & 1;
    if (issuer) {
      if (s + 1 < nsteps) {
        tdm_load_2d((uint32_t)(uintptr_t)&atile[p ^ 1][0],
                    A + (size_t)m0 * lda + (k0 + 32), (uint32_t)lda, 32u, 16u);
        __builtin_amdgcn_s_wait_tensorcnt(1);    // tile s landed, s+1 in flight
      } else {
        __builtin_amdgcn_s_wait_tensorcnt(0);
      }
    }
    __syncthreads();                             // tile p visible to all waves
    if (active) {
      v16bf a = load_a_frag(&atile[p][0], 32, 0, 0);
#pragma unroll
      for (int c = 0; c < NACC; ++c) {
        v16bf b = BT ? load_bt_frag(Bm, ldb, n0 + 16 * c, k0)
                     : load_bn_frag(Bm, ldb, n0 + 16 * c, k0);
        acc[c] = wmma_bf16(a, b, acc[c]);
      }
    }
    __syncthreads();                             // reads done before overwrite
  }
#else
  if (active) {
    for (int k0 = 0; k0 < Ksize; k0 += 32) {
      v16bf a = load_a_frag(A, lda, m0, k0);
#pragma unroll
      for (int c = 0; c < NACC; ++c) {
        v16bf b = BT ? load_bt_frag(Bm, ldb, n0 + 16 * c, k0)
                     : load_bn_frag(Bm, ldb, n0 + 16 * c, k0);
        acc[c] = wmma_bf16(a, b, acc[c]);
      }
    }
  }
#endif
  if (active) {
#pragma unroll
    for (int c = 0; c < NACC; ++c)
      store_tile(acc[c], Cf, Cb, ldc, m0, n0 + 16 * c, bias, alpha);
  }
}

// ------------------------------------------------------ dilated conv as GEMM
// y[n,o] = sum_k sum_c X[n + (k-3)*dil, c] * Wk[o,c]  (zero outside batch).
// Left/right edges need zero-fill before the tensor start, so A stays on the
// direct predicated-load path. 4 accumulators per wave reuse each A fragment.
__global__ void conv_gemm(const unsigned short* __restrict__ X,
                          const unsigned short* __restrict__ Wd,  // [7][D][D] bf16
                          const float* __restrict__ bias,
                          unsigned short* __restrict__ ms, int dil, int colOff) {
  int wave = threadIdx.x >> 5;
  int m0 = blockIdx.y * 16;
  int n0 = (blockIdx.x * 4 + wave) * 64;
  v8f acc[4];
#pragma unroll
  for (int c = 0; c < 4; ++c) acc[c] = v8f{};
#pragma unroll
  for (int k = 0; k < KCONV; ++k) {
    int shift = (k - 3) * dil;
    const unsigned short* Wk = Wd + (size_t)k * D * D;
    for (int k0 = 0; k0 < D; k0 += 32) {
      v16bf a = load_a_frag_shift(X, m0, k0, shift);
#pragma unroll
      for (int c = 0; c < 4; ++c) {
        v16bf b = load_bt_frag(Wk, D, n0 + 16 * c, k0);
        acc[c] = wmma_bf16(a, b, acc[c]);
      }
    }
  }
#pragma unroll
  for (int c = 0; c < 4; ++c)
    store_tile(acc[c], nullptr, ms + colOff, 5 * D, m0, n0 + 16 * c, bias, 1.f);
}

// -------------------------------------------------- GAT aggregation (flash-style)
// hp[i, h*128+d] = sum_j exp(leaky(e1[i,h]+e2[j,h]) - rowmax) * hg[j, h*128+d]
// rowmax = leaky(e1[i,h] + max_j e2[j,h])  (exact: leaky_relu is monotonic)
__global__ void gat_aggregate(const float* __restrict__ e1, const float* __restrict__ e2,
                              const float* __restrict__ m2,
                              const unsigned short* __restrict__ hg,
                              float* __restrict__ hp) {
  int h = blockIdx.z;
  int wave = threadIdx.x >> 5;
  int m0 = blockIdx.y * 16;
  int n0 = (blockIdx.x * 4 + wave) * 32;
  if (n0 >= DG) return;
  int lane = threadIdx.x & 31;
  int m = lane & 15, g = lane >> 4;
  float e1v = e1[(m0 + m) * HG + h];
  float rm = lrelu(e1v + m2[h]);
  v8f acc0 = {}; v8f acc1 = {};
  for (int j0 = 0; j0 < NN; j0 += 32) {
    union { v16bf v; uint32_t u[8]; } af;
#pragma unroll
    for (int i = 0; i < 4; ++i) {
      int k = 8 * g + 2 * i;
      float p0 = __expf(lrelu(e1v + e2[(j0 + k) * HG + h]) - rm);
      float p1 = __expf(lrelu(e1v + e2[(j0 + k + 1) * HG + h]) - rm);
      af.u[i] = pack2bf(p0, p1);
      int k2 = 16 + k;
      float q0 = __expf(lrelu(e1v + e2[(j0 + k2) * HG + h]) - rm);
      float q1 = __expf(lrelu(e1v + e2[(j0 + k2 + 1) * HG + h]) - rm);
      af.u[i + 4] = pack2bf(q0, q1);
    }
    v16bf b0 = load_bn_frag(hg, D, h * DG + n0, j0);
    v16bf b1 = load_bn_frag(hg, D, h * DG + n0 + 16, j0);
    acc0 = wmma_bf16(af.v, b0, acc0);
    acc1 = wmma_bf16(af.v, b1, acc1);
  }
  store_tile(acc0, hp + h * DG, nullptr, D, m0, n0, nullptr, 1.f);
  store_tile(acc1, hp + h * DG, nullptr, D, m0, n0 + 16, nullptr, 1.f);
}

// ------------------------------------------------------------- small kernels
__global__ void cast_f32_bf16(const float* __restrict__ s, unsigned short* __restrict__ d, int n) {
  int i = blockIdx.x * blockDim.x + threadIdx.x;
  if (i < n) d[i] = f2bf(s[i]);
}

// conv_w (o,c,k) f32 -> [k][o][c] bf16 (so each tap is a W row-major GEMM operand)
__global__ void repack_conv(const float* __restrict__ w, unsigned short* __restrict__ out) {
  int i = blockIdx.x * blockDim.x + threadIdx.x;
  if (i >= D * D * KCONV) return;
  int k = i % KCONV;
  int c = (i / KCONV) % D;
  int o = i / (KCONV * D);
  out[(size_t)k * D * D + (size_t)o * D + c] = f2bf(w[i]);
}

__global__ void ln_gelu(const float* __restrict__ zp, const float* __restrict__ g,
                        const float* __restrict__ b, unsigned short* __restrict__ zbf) {
  int row = blockIdx.x;
  const float* x = zp + (size_t)row * D;
  __shared__ float red[256];
  float s = 0.f;
  for (int i = threadIdx.x; i < D; i += 256) s += x[i];
  red[threadIdx.x] = s; __syncthreads();
  for (int o = 128; o > 0; o >>= 1) { if (threadIdx.x < o) red[threadIdx.x] += red[threadIdx.x + o]; __syncthreads(); }
  float mu = red[0] / D; __syncthreads();
  float v = 0.f;
  for (int i = threadIdx.x; i < D; i += 256) { float d = x[i] - mu; v += d * d; }
  red[threadIdx.x] = v; __syncthreads();
  for (int o = 128; o > 0; o >>= 1) { if (threadIdx.x < o) red[threadIdx.x] += red[threadIdx.x + o]; __syncthreads(); }
  float rstd = rsqrtf(red[0] / D + 1e-5f);
  for (int i = threadIdx.x; i < D; i += 256) {
    float y = (x[i] - mu) * rstd * g[i] + b[i];
    y = 0.5f * y * (1.f + erff(y * 0.70710678118f));     // exact GELU
    zbf[(size_t)row * D + i] = f2bf(y);
  }
}

__global__ void softmax256(const float* __restrict__ S, unsigned short* __restrict__ P) {
  int row = blockIdx.x;
  int t = threadIdx.x;
  float v = S[(size_t)row * 256 + t];
  __shared__ float red[256];
  red[t] = v; __syncthreads();
  for (int o = 128; o > 0; o >>= 1) { if (t < o) red[t] = fmaxf(red[t], red[t + o]); __syncthreads(); }
  float mx = red[0]; __syncthreads();
  float e = __expf(v - mx);
  red[t] = e; __syncthreads();
  for (int o = 128; o > 0; o >>= 1) { if (t < o) red[t] += red[t + o]; __syncthreads(); }
  P[(size_t)row * 256 + t] = f2bf(e / red[0]);
}

__global__ void gat_scores(const unsigned short* __restrict__ hg,
                           const float* __restrict__ a1, const float* __restrict__ a2,
                           float* __restrict__ e1, float* __restrict__ e2) {
  int idx = blockIdx.x * blockDim.x + threadIdx.x;   // over N*HG
  if (idx >= NN * HG) return;
  int n = idx / HG, h = idx % HG;
  const unsigned short* p = hg + (size_t)n * D + h * DG;
  float s1 = 0.f, s2 = 0.f;
  for (int d = 0; d < DG; ++d) { float v = bf2f(p[d]); s1 += v * a1[d]; s2 += v * a2[d]; }
  e1[idx] = s1; e2[idx] = s2;
}

__global__ void gat_colmax(const float* __restrict__ e2, float* __restrict__ m2) {
  int h = blockIdx.x;
  __shared__ float red[256];
  float m = -1e30f;
  for (int j = threadIdx.x; j < NN; j += 256) m = fmaxf(m, e2[j * HG + h]);
  red[threadIdx.x] = m; __syncthreads();
  for (int o = 128; o > 0; o >>= 1) { if (threadIdx.x < o) red[threadIdx.x] = fmaxf(red[threadIdx.x], red[threadIdx.x + o]); __syncthreads(); }
  if (threadIdx.x == 0) m2[h] = red[0];
}

__global__ void gat_denom(const float* __restrict__ e1, const float* __restrict__ e2,
                          const float* __restrict__ m2, float* __restrict__ zinv) {
  int i = blockIdx.x, h = blockIdx.y;
  float e1v = e1[i * HG + h];
  float rm = lrelu(e1v + m2[h]);
  __shared__ float red[256];
  float s = 0.f;
  for (int j = threadIdx.x; j < NN; j += 256)
    s += __expf(lrelu(e1v + e2[j * HG + h]) - rm);
  red[threadIdx.x] = s; __syncthreads();
  for (int o = 128; o > 0; o >>= 1) { if (threadIdx.x < o) red[threadIdx.x] += red[threadIdx.x + o]; __syncthreads(); }
  if (threadIdx.x == 0) zinv[i * HG + h] = 1.f / red[0];
}

__global__ void gat_finish(const float* __restrict__ hp, const float* __restrict__ zinv,
                           const float* __restrict__ feats, float* __restrict__ out) {
  int i = blockIdx.x * blockDim.x + threadIdx.x;   // over N*D
  if (i >= NN * D) return;
  int n = i / D, c = i % D, h = c / DG;
  float v = hp[i] * zinv[n * HG + h];
  v = v > 0.f ? v : (__expf(v) - 1.f);             // ELU
  out[i] = v + feats[i];
}

// ------------------------------------------------------------------- driver
extern "C" void kernel_launch(void* const* d_in, const int* in_sizes, int n_in,
                              void* d_out, int out_size, void* d_ws, size_t ws_size,
                              hipStream_t stream) {
  (void)in_sizes; (void)n_in; (void)out_size; (void)ws_size;
  const float* features = (const float*)d_in[0];
  const float* conv_w[5], *conv_b[5];
  for (int i = 0; i < 5; ++i) { conv_w[i] = (const float*)d_in[1 + 2 * i]; conv_b[i] = (const float*)d_in[2 + 2 * i]; }
  const float* fp_w = (const float*)d_in[11]; const float* fp_b = (const float*)d_in[12];
  const float* ln_g = (const float*)d_in[13]; const float* ln_b = (const float*)d_in[14];
  const float* wq = (const float*)d_in[15];  const float* bq = (const float*)d_in[16];
  const float* wk = (const float*)d_in[17];  const float* bk = (const float*)d_in[18];
  const float* wv = (const float*)d_in[19];  const float* bv = (const float*)d_in[20];
  const float* wo = (const float*)d_in[21];  const float* bo = (const float*)d_in[22];
  const float* gat_w = (const float*)d_in[23]; const float* gat_b = (const float*)d_in[24];
  const float* gat_a1 = (const float*)d_in[25]; const float* gat_a2 = (const float*)d_in[26];
  float* out = (float*)d_out;

  char* p = (char*)d_ws;
  auto alloc = [&](size_t bytes) -> void* {
    void* r = (void*)p; p += (bytes + 255) & ~(size_t)255; return r;
  };
  unsigned short* xbf = (unsigned short*)alloc((size_t)NN * D * 2);
  unsigned short* cw[5];
  for (int i = 0; i < 5; ++i) cw[i] = (unsigned short*)alloc((size_t)KCONV * D * D * 2);
  unsigned short* fpw = (unsigned short*)alloc((size_t)D * 5 * D * 2);
  unsigned short* wqb = (unsigned short*)alloc((size_t)D * D * 2);
  unsigned short* wkb = (unsigned short*)alloc((size_t)D * D * 2);
  unsigned short* wvb = (unsigned short*)alloc((size_t)D * D * 2);
  unsigned short* wob = (unsigned short*)alloc((size_t)D * D * 2);
  unsigned short* gwb = (unsigned short*)alloc((size_t)D * D * 2);
  unsigned short* ms  = (unsigned short*)alloc((size_t)NN * 5 * D * 2);
  float*          zpre = (float*)alloc((size_t)NN * D * 4);
  unsigned short* zbf = (unsigned short*)alloc((size_t)NN * D * 2);
  unsigned short* qb  = (unsigned short*)alloc((size_t)NN * D * 2);
  unsigned short* kb  = (unsigned short*)alloc((size_t)NN * D * 2);
  unsigned short* vb  = (unsigned short*)alloc((size_t)NN * D * 2);
  float*          scores = (float*)alloc((size_t)BB * HM * TT * TT * 4);
  unsigned short* Pbf = (unsigned short*)alloc((size_t)BB * HM * TT * TT * 2);
  unsigned short* ctxb = (unsigned short*)alloc((size_t)NN * D * 2);
  unsigned short* attb = (unsigned short*)alloc((size_t)NN * D * 2);
  unsigned short* hgb  = (unsigned short*)alloc((size_t)NN * D * 2);
  float* e1v  = (float*)alloc((size_t)NN * HG * 4);
  float* e2v  = (float*)alloc((size_t)NN * HG * 4);
  float* m2v  = (float*)alloc(32 * 4);
  float* zinv = (float*)alloc((size_t)NN * HG * 4);
  float* hp   = (float*)alloc((size_t)NN * D * 4);

  dim3 blk(128);
  // 1. precision conversion / weight repack
  cast_f32_bf16<<<(NN * D + 255) / 256, 256, 0, stream>>>(features, xbf, NN * D);
  for (int i = 0; i < 5; ++i)
    repack_conv<<<(D * D * KCONV + 255) / 256, 256, 0, stream>>>(conv_w[i], cw[i]);
  cast_f32_bf16<<<(D * 5 * D + 255) / 256, 256, 0, stream>>>(fp_w, fpw, D * 5 * D);
  cast_f32_bf16<<<(D * D + 255) / 256, 256, 0, stream>>>(wq, wqb, D * D);
  cast_f32_bf16<<<(D * D + 255) / 256, 256, 0, stream>>>(wk, wkb, D * D);
  cast_f32_bf16<<<(D * D + 255) / 256, 256, 0, stream>>>(wv, wvb, D * D);
  cast_f32_bf16<<<(D * D + 255) / 256, 256, 0, stream>>>(wo, wob, D * D);
  cast_f32_bf16<<<(D * D + 255) / 256, 256, 0, stream>>>(gat_w, gwb, D * D);

  // 2. five dilated convs -> ms (N x 5D, bf16, bias folded); 256 cols/block
  const int dils[5] = {1, 2, 4, 8, 16};
  for (int i = 0; i < 5; ++i)
    conv_gemm<<<dim3(4, NN / 16), blk, 0, stream>>>(xbf, cw[i], conv_b[i], ms, dils[i], i * D);

  // 3. fusion projection (2048x5120)*(5120x1024) -> zpre f32
  gemm_wmma<true, 4><<<dim3(4, NN / 16, 1), blk, 0, stream>>>(
      ms, 5 * D, fpw, 5 * D, zpre, nullptr, D, fp_b, 1.f, 5 * D, D, 1, 0, 0, 0, 0, 0, 0);

  // 4. LayerNorm + exact GELU -> zbf
  ln_gelu<<<NN, 256, 0, stream>>>(zpre, ln_g, ln_b, zbf);

  // 5. Q/K/V projections
  gemm_wmma<true, 4><<<dim3(4, NN / 16, 1), blk, 0, stream>>>(
      zbf, D, wqb, D, nullptr, qb, D, bq, 1.f, D, D, 1, 0, 0, 0, 0, 0, 0);
  gemm_wmma<true, 4><<<dim3(4, NN / 16, 1), blk, 0, stream>>>(
      zbf, D, wkb, D, nullptr, kb, D, bk, 1.f, D, D, 1, 0, 0, 0, 0, 0, 0);
  gemm_wmma<true, 4><<<dim3(4, NN / 16, 1), blk, 0, stream>>>(
      zbf, D, wvb, D, nullptr, vb, D, bv, 1.f, D, D, 1, 0, 0, 0, 0, 0, 0);

  // 6. scores[b,h] = Q_h K_h^T / 8  (z = b*16+h; one block covers 256 cols)
  gemm_wmma<true, 4><<<dim3(1, TT / 16, BB * HM), blk, 0, stream>>>(
      qb, D, kb, D, scores, nullptr, TT, nullptr, 0.125f, DH, TT, HM,
      (long)TT * D, DH, (long)TT * D, DH, (long)HM * TT * TT, (long)TT * TT);

  // 7. softmax rows -> Pbf
  softmax256<<<BB * HM * TT, 256, 0, stream>>>(scores, Pbf);

  // 8. ctx[b,h] = P V_h (B row-major); 64-thread blocks => both waves active
  gemm_wmma<false, 2><<<dim3(1, TT / 16, BB * HM), dim3(64), 0, stream>>>(
      Pbf, TT, vb, D, nullptr, ctxb, D, nullptr, 1.f, TT, DH, HM,
      (long)HM * TT * TT, (long)TT * TT, (long)TT * D, DH, (long)TT * D, DH);

  // 9. output projection + 10. GAT projection
  gemm_wmma<true, 4><<<dim3(4, NN / 16, 1), blk, 0, stream>>>(
      ctxb, D, wob, D, nullptr, attb, D, bo, 1.f, D, D, 1, 0, 0, 0, 0, 0, 0);
  gemm_wmma<true, 4><<<dim3(4, NN / 16, 1), blk, 0, stream>>>(
      attb, D, gwb, D, nullptr, hgb, D, gat_b, 1.f, D, D, 1, 0, 0, 0, 0, 0, 0);

  // 11. GAT attention logits, exact row-max (monotone leaky), denominators
  gat_scores<<<(NN * HG + 255) / 256, 256, 0, stream>>>(hgb, gat_a1, gat_a2, e1v, e2v);
  gat_colmax<<<HG, 256, 0, stream>>>(e2v, m2v);
  gat_denom<<<dim3(NN, HG), 256, 0, stream>>>(e1v, e2v, m2v, zinv);

  // 12. flash-style GAT aggregation (alpha never materialized) -> hp
  gat_aggregate<<<dim3(1, NN / 16, HG), blk, 0, stream>>>(e1v, e2v, m2v, hgb, hp);

  // 13. normalize, ELU, residual -> out f32
  gat_finish<<<(NN * D + 255) / 256, 256, 0, stream>>>(hp, zinv, features, out);
}
